// AxCoreTypedLinearFP16_43602507989106
// MI455X (gfx1250) — matrix operand order repaired and checked by reading, not compiled, not run on hardware
//
#include <hip/hip_runtime.h>
#include <stdint.h>

typedef _Float16 h8   __attribute__((ext_vector_type(8)));
typedef _Float16 v16h __attribute__((ext_vector_type(16)));
typedef float    v8f  __attribute__((ext_vector_type(8)));

#define MBLK 128
#define NBLK 256
#define KBLK 128                         // == scale group size G
#define SH   136                         // LDS row stride in halves (128 + 8 pad)
#define TILE_A_HALVES (128 * SH)         // x tile: 128 rows
#define TILE_B_HALVES (256 * SH)         // W tile: 256 rows
#define STAGE_HALVES (TILE_A_HALVES + TILE_B_HALVES)

// Copy ROWS x 256-byte (128 fp16) tile global->LDS via CDNA5 async path.
// ROWS*16 chunks of 16B spread over 256 threads.
template <int ROWS>
__device__ __forceinline__ void async_copy_tile(const _Float16* gbase, int ld_elems,
                                                uint32_t lds_base, int tid) {
#pragma unroll
  for (int i = 0; i < ROWS / 16; ++i) {
    const int c   = tid + i * 256;
    const int row = c >> 4;              // 16 chunks per row
    const int col = c & 15;
    uint64_t ga = (uint64_t)(uintptr_t)(gbase + (size_t)row * ld_elems) + (uint32_t)(col * 16);
    uint32_t la = lds_base + (uint32_t)(row * (SH * 2) + col * 16);
    asm volatile("global_load_async_to_lds_b128 %0, %1, off"
                 :: "v"(la), "v"(ga) : "memory");
  }
}

__global__ __launch_bounds__(256)
void AxCoreTypedLinearFP16_kernel(const _Float16* __restrict__ x,
                                  const _Float16* __restrict__ w,
                                  const _Float16* __restrict__ scales,
                                  const _Float16* __restrict__ bias,
                                  _Float16* __restrict__ y,
                                  int M, int N, int K, int ngroups) {
  __shared__ _Float16 smem[2 * STAGE_HALVES];   // double-buffered A + W (208,896 B)

  const int tid  = threadIdx.x;
  const int lane = tid & 31;
  const int wave = tid >> 5;        // 8 waves
  const int wm   = wave & 1;        // 2 waves along M (64 rows each)
  const int wn   = wave >> 1;       // 4 waves along N (64 cols each)
  const int lr   = lane & 15;       // fragment row / column index
  const int lh   = lane >> 4;       // half-wave select (K sub-block)

  const int m0 = blockIdx.y * MBLK;
  const int n0 = blockIdx.x * NBLK;

  const uint32_t lds0 = (uint32_t)(uintptr_t)(&smem[0]);
  const int nkb = K / KBLK;

  const _Float16* xg = x + (size_t)m0 * K;
  const _Float16* wg = w + (size_t)n0 * K;

  // scale row base for this wave's 4 column tiles (lane <-> column in B layout)
  const _Float16* srow[4];
#pragma unroll
  for (int nf = 0; nf < 4; ++nf)
    srow[nf] = scales + (size_t)(n0 + wn * 64 + nf * 16 + lr) * ngroups;

  v8f acc[4][4] = {};               // 64x64 wave tile: 16 accumulator tiles

  // prologue: stage 0 into buffer 0
  async_copy_tile<128>(xg, K, lds0, tid);
  async_copy_tile<256>(wg, K, lds0 + TILE_A_HALVES * 2, tid);

  for (int kb = 0; kb < nkb; ++kb) {
    // per-(column, k-group) dequant scales: global loads, independent of LDS
    // state -> issue before the async wait so their latency is hidden
    _Float16 sc[4];
#pragma unroll
    for (int nf = 0; nf < 4; ++nf) sc[nf] = srow[nf][kb];

    // prefetch next stage into the other buffer; the barrier at the end of
    // iteration kb-1 guarantees no wave is still reading it
    if (kb + 1 < nkb) {
      const uint32_t nb = lds0 + (uint32_t)(((kb + 1) & 1) * (STAGE_HALVES * 2));
      async_copy_tile<128>(xg + (kb + 1) * KBLK, K, nb, tid);
      async_copy_tile<256>(wg + (kb + 1) * KBLK, K, nb + TILE_A_HALVES * 2, tid);
      asm volatile("s_wait_asynccnt 24" ::: "memory");  // 24 older ops (this stage) done
    } else {
      asm volatile("s_wait_asynccnt 0" ::: "memory");
    }
    __syncthreads();   // make all waves' async data visible

    const _Float16* As = smem + (size_t)(kb & 1) * STAGE_HALVES;
    const _Float16* Bs = As + TILE_A_HALVES;

#pragma unroll
    for (int ks = 0; ks < 4; ++ks) {         // 4 x (k=32) per 128-group
      // Load all 4 A fragments for this k-step (reused by all 4 nf groups).
      v16h a[4];
#pragma unroll
      for (int mf = 0; mf < 4; ++mf) {
        // A 16x32 fp16 layout: lane lr = row; halves {k0..7, k16..23} (+8 upper half-wave)
        const _Float16* ap = As + (wm * 64 + mf * 16 + lr) * SH + ks * 32 + lh * 8;
        h8 lo = *(const h8*)ap;
        h8 hi = *(const h8*)(ap + 16);
        a[mf] = __builtin_shufflevector(lo, hi, 0,1,2,3,4,5,6,7,8,9,10,11,12,13,14,15);
      }
      // One B fragment live at a time -> low register pressure, deep ds prefetch.
#pragma unroll
      for (int nf = 0; nf < 4; ++nf) {
        // B 32x16 fp16 layout: lane lr = column (== W row); halves {k0..7, k16..23} (+8 upper half-wave)
        const _Float16* bp = Bs + (wn * 64 + nf * 16 + lr) * SH + ks * 32 + lh * 8;
        h8 b0 = *(const h8*)bp;
        h8 b1 = *(const h8*)(bp + 16);
        v16h braw = __builtin_shufflevector(b0, b1, 0,1,2,3,4,5,6,7,8,9,10,11,12,13,14,15);
        v16h sv;
#pragma unroll
        for (int i = 0; i < 16; ++i) sv[i] = sc[nf];
        const v16h b = braw * sv;   // fp16 dequant (v_pk_mul_f16 x8) == reference rounding
#pragma unroll
        for (int mf = 0; mf < 4; ++mf)   // shared B, disjoint acc -> back-to-back WMMAs
          acc[mf][nf] = __builtin_amdgcn_wmma_f32_16x16x32_f16(
              false, a[mf], false, b, (short)0, acc[mf][nf], false, false);
      }
    }
    __syncthreads();   // all waves done reading this buffer before it is refilled
  }

  // epilogue: bias (per-lane scalar since lane <-> output column) + fp16 store
#pragma unroll
  for (int nf = 0; nf < 4; ++nf) {
    const int n = n0 + wn * 64 + nf * 16 + lr;
    const float bf = (float)bias[n];
#pragma unroll
    for (int mf = 0; mf < 4; ++mf) {
      const int m = m0 + wm * 64 + mf * 16 + lh * 8;
#pragma unroll
      for (int r = 0; r < 8; ++r) {
        y[(size_t)(m + r) * N + n] = (_Float16)(acc[mf][nf][r] + bf);
      }
    }
  }
}

extern "C" void kernel_launch(void* const* d_in, const int* in_sizes, int n_in,
                              void* d_out, int out_size, void* d_ws, size_t ws_size,
                              hipStream_t stream) {
  const _Float16* x      = (const _Float16*)d_in[0];
  const _Float16* w      = (const _Float16*)d_in[1];
  const _Float16* scales = (const _Float16*)d_in[2];
  const _Float16* bias   = (const _Float16*)d_in[3];
  // d_in[4] (types) only selects approximation modes; exact reference ignores it.
  _Float16* y = (_Float16*)d_out;

  const int O = in_sizes[3];                 // 4096
  const int I = in_sizes[1] / O;             // 4096
  const int M = in_sizes[0] / I;             // 8192
  const int ngroups = in_sizes[2] / O;       // I / G = 32  (G == 128 == KBLK)

  dim3 grid(O / NBLK, M / MBLK);             // (16, 64)
  AxCoreTypedLinearFP16_kernel<<<grid, dim3(256), 0, stream>>>(
      x, w, scales, bias, y, M, O, I, ngroups);
}
